// PtrScorer_77189152244460
// MI455X (gfx1250) — matrix operand order repaired and checked by reading, not compiled, not run on hardware
//
#include <hip/hip_runtime.h>

// ---------------------------------------------------------------------------
// Types for CDNA5 WMMA (wave32): bf16 A/B, f32 accumulate, 16x16x32
// ---------------------------------------------------------------------------
typedef __attribute__((ext_vector_type(8)))  __bf16 v8bf;
typedef __attribute__((ext_vector_type(16))) __bf16 v16bf;
typedef __attribute__((ext_vector_type(8)))  float  v8f;

#define BATCH 32
#define SLEN  2048
#define DIM   512
#define HDIM  512
#define G4H   2048   // 4*H
#define NPRED 8
#define NEG_INF -1e9f
#define RT    8      // attention tile rows
#define NT    (SLEN / RT)

#if __has_builtin(__builtin_amdgcn_tensor_load_to_lds)
#define USE_TDM 1
#endif

__device__ __forceinline__ __bf16 to_bf16(float f) {
  unsigned int u = __builtin_bit_cast(unsigned int, f);
  unsigned int r = u + 0x7FFFu + ((u >> 16) & 1u);   // round-to-nearest-even
  unsigned short s = (unsigned short)(r >> 16);
  return __builtin_bit_cast(__bf16, s);
}

__device__ __forceinline__ float fsigmoid(float x) {
  return 1.0f / (1.0f + __expf(-x));
}

// ---------------------------------------------------------------------------
// WMMA tile loaders (ISA 7.12.2, wave32)
// A 16x32 bf16: lane = m + 16*h -> K chunks [8h..8h+7] and [16+8h..16+8h+7]
// ---------------------------------------------------------------------------
__device__ __forceinline__ v16bf load_a_tile(const __bf16* A, int lda, int m0, int k0) {
  int lane = threadIdx.x & 31;
  int m  = m0 + (lane & 15);
  int kh = (lane >> 4) * 8;
  const __bf16* p = A + (size_t)m * lda + k0 + kh;
  v8bf lo = *(const v8bf*)(p);        // K = 8h .. 8h+7
  v8bf hi = *(const v8bf*)(p + 16);   // K = 16+8h .. 16+8h+7
  v16bf r;
#pragma unroll
  for (int i = 0; i < 8; ++i) { r[i] = lo[i]; r[8 + i] = hi[i]; }
  return r;
}

// B 32x16 bf16 from row-major W[N,K]: lane = n + 16*h -> K chunks [8h..8h+7] and [16+8h..16+8h+7]
__device__ __forceinline__ v16bf load_b_tile(const __bf16* B, int ldb, int n0, int k0) {
  int lane = threadIdx.x & 31;
  int n  = n0 + (lane & 15);
  int kh = (lane >> 4) * 8;
  const __bf16* p = B + (size_t)n * ldb + k0 + kh;
  v8bf lo = *(const v8bf*)(p);
  v8bf hi = *(const v8bf*)(p + 16);
  v16bf r;
#pragma unroll
  for (int i = 0; i < 8; ++i) { r[i] = lo[i]; r[8 + i] = hi[i]; }
  return r;
}

// ---------------------------------------------------------------------------
// TDM: 2D tile load Global -> LDS (D# per CDNA5 ISA)
// tile = rows x DIM f32, row stride DIM elements. Issued once by one wave.
// This toolchain exposes the 6-arg builtin:
//   (u32x4 g0, i32x8 g1, i32x4 g2, i32x4 g3, i32x8, i32 cpol)
// ---------------------------------------------------------------------------
#ifdef USE_TDM
typedef unsigned int u32x4 __attribute__((ext_vector_type(4)));
typedef int i32x8 __attribute__((ext_vector_type(8)));
typedef int i32x4 __attribute__((ext_vector_type(4)));

__device__ __forceinline__ void tdm_load_tile(const float* gsrc, float* lds_dst, int rows) {
  unsigned long long ga = (unsigned long long)(uintptr_t)gsrc;
  unsigned int lds = (unsigned int)(uintptr_t)lds_dst;  // low 32 bits = LDS byte addr
  u32x4 g0;
  g0[0] = 1u;                                            // count=1, user descriptor
  g0[1] = lds;                                           // lds_addr [63:32]
  g0[2] = (unsigned int)(ga & 0xFFFFFFFFu);              // global_addr[31:0]
  g0[3] = (unsigned int)((ga >> 32) & 0x01FFFFFFu)       // global_addr[56:32]
          | (2u << 30);                                  // type=2 ("image")
  i32x8 g1;
  g1[0] = (int)(2u << 16);                               // data_size=4B; wg_mask=0
  g1[1] = (int)((DIM & 0xFFFF) << 16);                   // tensor_dim0[15:0] @ bits63:48
  g1[2] = (int)(((unsigned)DIM >> 16) |                  // tensor_dim0[31:16]
                ((SLEN & 0xFFFF) << 16));                // tensor_dim1[15:0]
  g1[3] = (int)(((unsigned)SLEN >> 16) |                 // tensor_dim1[31:16]
                ((DIM & 0xFFFF) << 16));                 // tile_dim0 @ bits127:112
  g1[4] = (int)(rows & 0xFFFF);                          // tile_dim1; tile_dim2=0
  g1[5] = (int)DIM;                                      // tensor_dim0_stride[31:0]
  g1[6] = 0;                                             // stride hi / dim1_stride lo
  g1[7] = 0;
  i32x4 z4 = {0, 0, 0, 0};                               // groups 2/3 unused (2D)
  i32x8 z8 = {0, 0, 0, 0, 0, 0, 0, 0};
  __builtin_amdgcn_tensor_load_to_lds(g0, g1, z4, z4, z8, 0);
}
#endif

// ---------------------------------------------------------------------------
// One-time conversion kernels
// ---------------------------------------------------------------------------
__global__ void cvt_f32_to_bf16(const float* __restrict__ src, __bf16* __restrict__ dst, int n) {
  int i = blockIdx.x * blockDim.x + threadIdx.x;
  if (i < n) dst[i] = to_bf16(src[i]);
}

// dst[n*K + k] = src[k*N + n]  (attn_Wq [H,D] row-major -> WqT [D,H] row-major)
__global__ void cvt_transpose_bf16(const float* __restrict__ src, __bf16* __restrict__ dst,
                                   int K, int N) {
  int i = blockIdx.x * blockDim.x + threadIdx.x;
  if (i < K * N) {
    int n = i / K, k = i - n * K;
    dst[i] = to_bf16(src[k * N + n]);
  }
}

__global__ void init_state(const float* __restrict__ init_h, const float* __restrict__ init_c,
                           const float* __restrict__ init_in,
                           float* __restrict__ c, __bf16* __restrict__ hb, __bf16* __restrict__ xb) {
  int idx = blockIdx.x * blockDim.x + threadIdx.x;  // BATCH*HDIM
  if (idx < BATCH * HDIM) {
    int j = idx & (HDIM - 1);
    c[idx]  = init_c[j];
    hb[idx] = to_bf16(init_h[j]);
    xb[idx] = to_bf16(init_in[j]);
  }
}

// ---------------------------------------------------------------------------
// gates[32,2048] = x_bf16 @ W_ih^T + h_bf16 @ W_hh^T + b_ih + b_hh  (WMMA)
// grid (G4H/16, 2), one wave per 16x16 C tile
// ---------------------------------------------------------------------------
__global__ void gates_wmma_kernel(const __bf16* __restrict__ xb, const __bf16* __restrict__ hb,
                                  const __bf16* __restrict__ Wih, const __bf16* __restrict__ Whh,
                                  const float* __restrict__ b_ih, const float* __restrict__ b_hh,
                                  float* __restrict__ gates) {
  int n0 = blockIdx.x * 16;
  int m0 = blockIdx.y * 16;
  v8f acc = {};
#pragma unroll 4
  for (int k0 = 0; k0 < DIM; k0 += 32) {
    v16bf a = load_a_tile(xb, DIM, m0, k0);
    v16bf b = load_b_tile(Wih, DIM, n0, k0);
    acc = __builtin_amdgcn_wmma_f32_16x16x32_bf16(false, a, false, b, (short)0, acc, false, false);
  }
#pragma unroll 4
  for (int k0 = 0; k0 < HDIM; k0 += 32) {
    v16bf a = load_a_tile(hb, HDIM, m0, k0);
    v16bf b = load_b_tile(Whh, HDIM, n0, k0);
    acc = __builtin_amdgcn_wmma_f32_16x16x32_bf16(false, a, false, b, (short)0, acc, false, false);
  }
  int lane = threadIdx.x & 31;
  int n  = n0 + (lane & 15);
  int mh = (lane >> 4) * 8;
  float bias = b_ih[n] + b_hh[n];
#pragma unroll
  for (int r = 0; r < 8; ++r)
    gates[(size_t)(m0 + mh + r) * G4H + n] = acc[r] + bias;
}

// ---------------------------------------------------------------------------
// q[32,512] = h_bf16 @ WqT^T  (WqT is [D,H] row-major = [N,K])  (WMMA)
// ---------------------------------------------------------------------------
__global__ void q_wmma_kernel(const __bf16* __restrict__ hb, const __bf16* __restrict__ WqT,
                              float* __restrict__ q) {
  int n0 = blockIdx.x * 16;
  int m0 = blockIdx.y * 16;
  v8f acc = {};
#pragma unroll 4
  for (int k0 = 0; k0 < HDIM; k0 += 32) {
    v16bf a = load_a_tile(hb, HDIM, m0, k0);
    v16bf b = load_b_tile(WqT, HDIM, n0, k0);
    acc = __builtin_amdgcn_wmma_f32_16x16x32_bf16(false, a, false, b, (short)0, acc, false, false);
  }
  int lane = threadIdx.x & 31;
  int n  = n0 + (lane & 15);
  int mh = (lane >> 4) * 8;
#pragma unroll
  for (int r = 0; r < 8; ++r)
    q[(size_t)(m0 + mh + r) * DIM + n] = acc[r];
}

// ---------------------------------------------------------------------------
// LSTM pointwise
// ---------------------------------------------------------------------------
__global__ void lstm_pointwise(const float* __restrict__ gates, float* __restrict__ c,
                               __bf16* __restrict__ hb) {
  int idx = blockIdx.x * blockDim.x + threadIdx.x;  // BATCH*HDIM
  if (idx >= BATCH * HDIM) return;
  int b = idx >> 9;
  int j = idx & (HDIM - 1);
  const float* g = gates + (size_t)b * G4H;
  float gi = fsigmoid(g[j]);
  float gf = fsigmoid(g[HDIM + j]);
  float gg = tanhf(g[2 * HDIM + j]);
  float go = fsigmoid(g[3 * HDIM + j]);
  float cn = gf * c[idx] + gi * gg;
  c[idx] = cn;
  hb[idx] = to_bf16(go * tanhf(cn));
}

// ---------------------------------------------------------------------------
// Single-pass flash-style attention per batch (256 threads / block):
// online softmax over enc tiles staged in LDS (TDM double-buffered when
// available). Scan stops at ceil(ns/RT) tiles: masked rows have exactly
// zero softmax weight (-1e9 logits), so they contribute nothing.
// ---------------------------------------------------------------------------
__global__ __launch_bounds__(256) void attn_kernel(
    const float* __restrict__ enc, const float* __restrict__ q,
    const int* __restrict__ num_sent,
    const float* __restrict__ score_W, const float* __restrict__ score_b,
    __bf16* __restrict__ xb, float* __restrict__ out_scores) {
  int b = blockIdx.x;
  int tid = threadIdx.x;
  int wave = tid >> 5, lane = tid & 31;

  __shared__ float tile[2][RT * DIM];   // 2 x 16 KB
  __shared__ float sq[DIM];             // 2 KB
  __shared__ float lt[RT];              // masked logits of current tile
  __shared__ float sw[RT];              // softmax weights of current tile
  __shared__ float red[256];

  const float* E = enc + (size_t)b * SLEN * DIM;
  for (int d = tid; d < DIM; d += 256) sq[d] = q[b * DIM + d];

  int ns = num_sent[b];
  int nt = (ns + RT - 1) / RT;
  if (nt > NT) nt = NT;

#ifdef USE_TDM
  if (wave == 0) tdm_load_tile(E, &tile[0][0], RT);   // prefetch tile 0
#endif

  float M = -3.4e38f, S = 0.f, o0 = 0.f, o1 = 0.f;
  int d0 = tid, d1 = tid + 256;

  for (int i = 0; i < nt; ++i) {
    float* T = &tile[i & 1][0];
#ifdef USE_TDM
    if (wave == 0) {
      if (i + 1 < nt) {
        tdm_load_tile(E + (size_t)(i + 1) * RT * DIM, &tile[(i + 1) & 1][0], RT);
        __builtin_amdgcn_s_wait_tensorcnt((short)1);  // in-order: tile i done
      } else {
        __builtin_amdgcn_s_wait_tensorcnt((short)0);
      }
    }
    __syncthreads();  // tile i visible to all waves
#else
    {
      const float4* g4 = (const float4*)(E + (size_t)i * RT * DIM);
      float4* t4 = (float4*)T;
      for (int e = tid; e < RT * DIM / 4; e += 256) t4[e] = g4[e];
    }
    __syncthreads();
#endif

    // logits: wave w computes row w (wave32 shuffle reduction, no barriers)
    {
      const float* row = T + wave * DIM;
      float acc = 0.f;
#pragma unroll
      for (int d = 0; d < DIM / 32; ++d) acc += row[lane + d * 32] * sq[lane + d * 32];
      for (int off = 16; off > 0; off >>= 1) acc += __shfl_down(acc, off, 32);
      if (lane == 0) {
        int s = i * RT + wave;
        lt[wave] = (s < ns) ? acc : NEG_INF;
      }
    }
    __syncthreads();

    // online softmax update
    float tmax = lt[0];
#pragma unroll
    for (int r = 1; r < RT; ++r) tmax = fmaxf(tmax, lt[r]);
    float newM = fmaxf(M, tmax);
    if (tid < RT) sw[tid] = __expf(lt[tid] - newM);
    __syncthreads();

    float scale = __expf(M - newM);
    float wsum = 0.f;
#pragma unroll
    for (int r = 0; r < RT; ++r) wsum += sw[r];
    S = S * scale + wsum;
    o0 *= scale; o1 *= scale;
#pragma unroll
    for (int r = 0; r < RT; ++r) {
      float w = sw[r];
      o0 += w * T[r * DIM + d0];
      o1 += w * T[r * DIM + d1];
    }
    M = newM;
    __syncthreads();  // everyone done with tile/sw before overwrite
  }

  float inv = 1.0f / S;
  o0 *= inv; o1 *= inv;
  xb[b * DIM + d0] = to_bf16(o0);
  xb[b * DIM + d1] = to_bf16(o1);

  // score = out . score_W + score_b
  red[tid] = o0 * score_W[d0] + o1 * score_W[d1];
  __syncthreads();
  for (int off = 128; off > 0; off >>= 1) {
    if (tid < off) red[tid] += red[tid + off];
    __syncthreads();
  }
  if (tid == 0) out_scores[b] = red[0] + score_b[0];
}

// ---------------------------------------------------------------------------
// Host launcher
// ---------------------------------------------------------------------------
extern "C" void kernel_launch(void* const* d_in, const int* in_sizes, int n_in,
                              void* d_out, int out_size, void* d_ws, size_t ws_size,
                              hipStream_t stream) {
  const float* enc     = (const float*)d_in[0];
  const int*   num_s   = (const int*)  d_in[1];
  // d_in[2] = num_pred (scalar == 8; scan length is static in the reference)
  const float* init_h  = (const float*)d_in[3];
  const float* init_c  = (const float*)d_in[4];
  const float* init_in = (const float*)d_in[5];
  const float* W_ih    = (const float*)d_in[6];
  const float* W_hh    = (const float*)d_in[7];
  const float* b_ih    = (const float*)d_in[8];
  const float* b_hh    = (const float*)d_in[9];
  const float* Wq      = (const float*)d_in[10];
  const float* score_W = (const float*)d_in[11];
  const float* score_b = (const float*)d_in[12];
  float* out = (float*)d_out;  // [NPRED, BATCH, 1]

  char* ws = (char*)d_ws;
  size_t off = 0;
  auto carve = [&](size_t bytes) -> void* {
    void* p = ws + off;
    off = (off + bytes + 255) & ~(size_t)255;
    return p;
  };
  __bf16* Wih_b = (__bf16*)carve((size_t)G4H * DIM * 2);   // 2 MB
  __bf16* Whh_b = (__bf16*)carve((size_t)G4H * HDIM * 2);  // 2 MB
  __bf16* WqT_b = (__bf16*)carve((size_t)DIM * HDIM * 2);  // 0.5 MB
  float*  gates = (float*) carve((size_t)BATCH * G4H * 4); // 256 KB
  float*  cbuf  = (float*) carve((size_t)BATCH * HDIM * 4);
  float*  qbuf  = (float*) carve((size_t)BATCH * DIM * 4);
  __bf16* hb    = (__bf16*)carve((size_t)BATCH * HDIM * 2);
  __bf16* xb    = (__bf16*)carve((size_t)BATCH * DIM * 2);

  {
    int n = G4H * DIM;
    cvt_f32_to_bf16<<<(n + 255) / 256, 256, 0, stream>>>(W_ih, Wih_b, n);
    cvt_f32_to_bf16<<<(n + 255) / 256, 256, 0, stream>>>(W_hh, Whh_b, n);
    int nq = DIM * HDIM;
    cvt_transpose_bf16<<<(nq + 255) / 256, 256, 0, stream>>>(Wq, WqT_b, HDIM, DIM);
    init_state<<<(BATCH * HDIM + 255) / 256, 256, 0, stream>>>(init_h, init_c, init_in,
                                                               cbuf, hb, xb);
  }

  for (int t = 0; t < NPRED; ++t) {
    gates_wmma_kernel<<<dim3(G4H / 16, BATCH / 16), 32, 0, stream>>>(
        xb, hb, Wih_b, Whh_b, b_ih, b_hh, gates);
    lstm_pointwise<<<(BATCH * HDIM + 255) / 256, 256, 0, stream>>>(gates, cbuf, hb);
    q_wmma_kernel<<<dim3(DIM / 16, BATCH / 16), 32, 0, stream>>>(hb, WqT_b, qbuf);
    attn_kernel<<<BATCH, 256, 0, stream>>>(enc, qbuf, num_s, score_W, score_b,
                                           xb, out + (size_t)t * BATCH);
  }
}
